// GraphUnet_60490319397368
// MI455X (gfx1250) — matrix operand order repaired and checked by reading, not compiled, not run on hardware
//
#include <hip/hip_runtime.h>
#include <hip/hip_bf16.h>
#include <math.h>

// ---------------- CDNA5 WMMA types ----------------
typedef __attribute__((ext_vector_type(16))) __bf16 v16bf;
typedef __attribute__((ext_vector_type(8)))  float  v8f;

#define GAT_ALPHA 0.2f
static constexpr int D = 320;     // hidden dim (= OUT_DIM); all GEMMs have dout == D
static constexpr int N0 = 4096;

__device__ __forceinline__ float lrelu(float x) { return x > 0.f ? x : GAT_ALPHA * x; }
__device__ __forceinline__ float elu1(float x)  { return x > 0.f ? x : (__expf(x) - 1.f); }

// permute k-index (swap bits 3<->4) so each lane's WMMA A-fragment is one
// contiguous 32B run in LDS (ISA 16-bit A 16x32 layout => per-lane k order
// [hi*8+0..7, 16+hi*8+0..7]).
__device__ __forceinline__ int kperm(int c) {
    return (c & 7) | ((c & 8) << 1) | ((c & 16) >> 1);
}

union Pack4 { __bf16 b[4]; unsigned long long u64; };
union Pack2 { __bf16 b[2]; unsigned int u32; };

// ---------------- small utility kernels ----------------
__global__ void init_v_kernel(float* v, int n) {
    int i = blockIdx.x * blockDim.x + threadIdx.x;
    if (i < n) v[i] = 1.0f / (float)n;
}

__global__ void zero_kernel(float* p, int total) {
    int i = blockIdx.x * blockDim.x + threadIdx.x;
    if (i < total) p[i] = 0.f;
}

// y = A @ v  (one wave per row, wave32 shfl reduction)
__global__ void gemv_kernel(const float* __restrict__ A, const float* __restrict__ v,
                            float* __restrict__ y, int n) {
    int row  = blockIdx.x * (blockDim.x >> 5) + (threadIdx.x >> 5);
    int lane = threadIdx.x & 31;
    if (row >= n) return;
    const float* ar = A + (size_t)row * n;
    float s = 0.f;
    for (int j = lane; j < n; j += 32) s += ar[j] * v[j];
    for (int off = 16; off > 0; off >>= 1) s += __shfl_xor(s, off, 32);
    if (lane == 0) y[row] = s;
}

// deterministic single-block sum of squares
__global__ void sumsq_kernel(const float* __restrict__ y, float* __restrict__ ss, int n) {
    __shared__ float sh[256];
    float s = 0.f;
    for (int i = threadIdx.x; i < n; i += 256) s += y[i] * y[i];
    sh[threadIdx.x] = s;
    __syncthreads();
    for (int st = 128; st > 0; st >>= 1) {
        if ((int)threadIdx.x < st) sh[threadIdx.x] += sh[threadIdx.x + st];
        __syncthreads();
    }
    if (threadIdx.x == 0) ss[0] = sh[0];
}

__global__ void normalize_kernel(const float* __restrict__ y, const float* __restrict__ ss,
                                 float* __restrict__ v, int n) {
    int i = blockIdx.x * blockDim.x + threadIdx.x;
    if (i < n) v[i] = y[i] / (sqrtf(ss[0]) + 1e-12f);
}

// Xc = concat([X (n x 319), centrality (n x 1)])
__global__ void build_xc_kernel(const float* __restrict__ X, const float* __restrict__ v,
                                float* __restrict__ Xc, int n) {
    int t = blockIdx.x * blockDim.x + threadIdx.x;
    if (t >= n * D) return;
    int i = t / D, c = t % D;
    Xc[t] = (c < D - 1) ? X[(size_t)i * (D - 1) + c] : v[i];
}

// ---------------- WMMA GEMM: C[n x 320] = A[n x k] @ B (+bias) ----------------
// k multiple of 32 (320 or 640). B row-major [k x 320], or (bT) B[c*k + gk].
// Block 256 threads = 8 waves; tile M=32, N=320; wave: M-half = w>>2,
// N-slice = (w&3)*80 (5 x 16-col WMMA tiles). Software pipelined: register
// prefetch of next tile + double-buffered LDS -> one barrier per K-step.
__global__ __launch_bounds__(256)
void wmma_gemm_kernel(const float* __restrict__ Ain, const float* __restrict__ Bin,
                      const float* __restrict__ bias, float* __restrict__ Cout,
                      int n, int k, int bT) {
    __shared__ alignas(64) __bf16 As[2][32 * 32];    // [m][k]
    __shared__ alignas(64) __bf16 BsT[2][D * 32];    // [ncol][k]
    const int row0  = blockIdx.x << 5;
    const int tid   = threadIdx.x;
    const int wave  = tid >> 5;
    const int lane  = tid & 31;
    const int mrow  = lane & 15;
    const int hi    = lane >> 4;
    const int mhalf = wave >> 2;
    const int ncol0 = (wave & 3) * 80;

    // hoisted staging coordinates
    const int ar = tid >> 3, ac = (tid & 7) << 2;
    int agr = row0 + ar; if (agr > n - 1) agr = n - 1;
    const float* aptr = Ain + (size_t)agr * k + ac;

    float4 fa;
    float4 fb[10];
    auto loadRegs = [&](int kq) {
        fa = *(const float4*)(aptr + kq);
        if (bT == 0) {
#pragma unroll
            for (int it = 0; it < 10; ++it) {
                int e = tid + (it << 8);
                int kr = e / 80, gc = (e % 80) << 2;
                fb[it] = *(const float4*)(Bin + (size_t)(kq + kr) * D + gc);
            }
        } else {
#pragma unroll
            for (int it = 0; it < 10; ++it) {
                int e = tid + (it << 8);
                int nr = e >> 3, gk = (e & 7) << 2;
                fb[it] = *(const float4*)(Bin + (size_t)nr * k + kq + gk);
            }
        }
    };
    auto storeTile = [&](int buf) {
        Pack4 p;
        p.b[0] = (__bf16)fa.x; p.b[1] = (__bf16)fa.y;
        p.b[2] = (__bf16)fa.z; p.b[3] = (__bf16)fa.w;
        *(unsigned long long*)(As[buf] + ar * 32 + ac) = p.u64;
        if (bT == 0) {
#pragma unroll
            for (int it = 0; it < 10; ++it) {
                int e = tid + (it << 8);
                int kr = e / 80, gc = (e % 80) << 2;
                BsT[buf][(gc + 0) * 32 + kr] = (__bf16)fb[it].x;
                BsT[buf][(gc + 1) * 32 + kr] = (__bf16)fb[it].y;
                BsT[buf][(gc + 2) * 32 + kr] = (__bf16)fb[it].z;
                BsT[buf][(gc + 3) * 32 + kr] = (__bf16)fb[it].w;
            }
        } else {
#pragma unroll
            for (int it = 0; it < 10; ++it) {
                int e = tid + (it << 8);
                int nr = e >> 3, gk = (e & 7) << 2;
                Pack4 q;
                q.b[0] = (__bf16)fb[it].x; q.b[1] = (__bf16)fb[it].y;
                q.b[2] = (__bf16)fb[it].z; q.b[3] = (__bf16)fb[it].w;
                *(unsigned long long*)(BsT[buf] + nr * 32 + gk) = q.u64;
            }
        }
    };

    v8f acc[5] = {};
    loadRegs(0);
    storeTile(0);
    for (int kk = 0; kk < k; kk += 32) {
        const int cur = (kk >> 5) & 1;
        __syncthreads();
        const bool more = (kk + 32 < k);
        if (more) loadRegs(kk + 32);                 // overlaps ds loads + WMMAs below
        const v16bf af = *(const v16bf*)(As[cur] + (mhalf * 16 + mrow) * 32 + hi * 16);
#pragma unroll
        for (int j = 0; j < 5; ++j) {
            const v16bf bfm = *(const v16bf*)(BsT[cur] + (ncol0 + j * 16 + mrow) * 32 + hi * 16);
            acc[j] = __builtin_amdgcn_wmma_f32_16x16x32_bf16(false, af, false, bfm,
                                                             (short)0, acc[j], false, false);
        }
        if (more) storeTile(cur ^ 1);                // opposite buffer: no extra barrier
    }
#pragma unroll
    for (int j = 0; j < 5; ++j) {
        int c = ncol0 + j * 16 + mrow;
        float bb = bias ? bias[c] : 0.f;
#pragma unroll
        for (int r = 0; r < 8; ++r) {
            int m = row0 + mhalf * 16 + r + hi * 8;
            if (m < n) Cout[(size_t)m * D + c] = acc[j][r] + bb;
        }
    }
}

// ---------------- GAT attention pieces ----------------
// Wh1 = Wh @ a[:D], Wh2 = Wh @ a[D:], one wave per row
__global__ void gemv2_kernel(const float* __restrict__ Wh, const float* __restrict__ avec,
                             float* __restrict__ Wh1, float* __restrict__ Wh2, int n) {
    int row  = blockIdx.x * (blockDim.x >> 5) + (threadIdx.x >> 5);
    int lane = threadIdx.x & 31;
    if (row >= n) return;
    const float* wr = Wh + (size_t)row * D;
    float s1 = 0.f, s2 = 0.f;
    for (int c = lane; c < D; c += 32) {
        float w = wr[c];
        s1 += w * avec[c];
        s2 += w * avec[D + c];
    }
    for (int off = 16; off > 0; off >>= 1) {
        s1 += __shfl_xor(s1, off, 32);
        s2 += __shfl_xor(s2, off, 32);
    }
    if (lane == 0) { Wh1[row] = s1; Wh2[row] = s2; }
}

// per-row masked-softmax stats (max, 1/Z, edge count); adjacency stays L2-resident
__global__ void attn_stats_kernel(const float* __restrict__ adj, const float* __restrict__ Wh1,
                                  const float* __restrict__ Wh2, float* __restrict__ rowM,
                                  float* __restrict__ rowZinv, int* __restrict__ rowCnt, int n) {
    int row  = blockIdx.x * (blockDim.x >> 5) + (threadIdx.x >> 5);
    int lane = threadIdx.x & 31;
    if (row >= n) return;
    const float* ar = adj + (size_t)row * n;
    float w1 = Wh1[row];
    float m = -INFINITY, s = 0.f;
    int cnt = 0;
    for (int j = lane; j < n; j += 32) {
        if (ar[j] > 0.f) {
            float e = lrelu(w1 + Wh2[j]);
            float mn = fmaxf(m, e);
            s = (s > 0.f ? s * __expf(m - mn) : 0.f) + __expf(e - mn);
            m = mn;
            cnt++;
        }
    }
    for (int off = 16; off > 0; off >>= 1) {
        float mo = __shfl_xor(m, off, 32);
        float so = __shfl_xor(s, off, 32);
        int   co = __shfl_xor(cnt, off, 32);
        float mn = fmaxf(m, mo);
        float sn = (s  > 0.f ? s  * __expf(m  - mn) : 0.f) +
                   (so > 0.f ? so * __expf(mo - mn) : 0.f);
        m = mn; s = sn; cnt += co;
    }
    if (lane == 0) {
        rowM[row]    = m;
        rowZinv[row] = (cnt > 0 && s > 0.f) ? 1.f / s : 0.f;
        rowCnt[row]  = cnt;
    }
}

// out[n x 320] = act( softmax_mask(P) @ Wh ) (+resid). Probability tile is
// generated flash-style from per-row stats (hoisted per block); same pipelined
// double-buffered structure as the GEMM. Adjacency traversed exactly once.
__global__ __launch_bounds__(256)
void attn_apply_kernel(const float* __restrict__ adj, const float* __restrict__ Wh,
                       const float* __restrict__ Wh1, const float* __restrict__ Wh2,
                       const float* __restrict__ rowM, const float* __restrict__ rowZinv,
                       const int* __restrict__ rowCnt, float* __restrict__ out,
                       const float* __restrict__ resid, int n, int act) {
    __shared__ alignas(64) __bf16 Ps[2][32 * 32];    // [m][k]
    __shared__ alignas(64) __bf16 BsT[2][D * 32];    // [ncol][k]
    const int row0  = blockIdx.x << 5;
    const int tid   = threadIdx.x;
    const int wave  = tid >> 5;
    const int lane  = tid & 31;
    const int mrow  = lane & 15;
    const int hi    = lane >> 4;
    const int mhalf = wave >> 2;
    const int ncol0 = (wave & 3) * 80;
    const float invn = 1.f / (float)n;

    // hoisted per-thread P-generation row state (independent of kk)
    int prow[2], pcol[2], pcnt[2];
    float pw1[2], prm[2], prz[2];
    const float* padj[2];
#pragma unroll
    for (int it = 0; it < 2; ++it) {
        int e = tid + (it << 8);
        prow[it] = e >> 4;
        pcol[it] = (e & 15) << 1;
        int i = row0 + prow[it]; if (i > n - 1) i = n - 1;
        pcnt[it] = rowCnt[i];
        pw1[it]  = Wh1[i];
        prm[it]  = rowM[i];
        prz[it]  = rowZinv[i];
        padj[it] = adj + (size_t)i * n;
    }

    float2 a2r[2], w2r[2];
    float4 fb[10];
    auto loadRegs = [&](int kq) {
#pragma unroll
        for (int it = 0; it < 2; ++it) {
            int j = kq + pcol[it];
            int jc = (j <= n - 2) ? j : (n - 2);     // n even, j even => j<n <=> j<=n-2
            a2r[it] = *(const float2*)(padj[it] + jc);
            w2r[it] = *(const float2*)(Wh2 + jc);
        }
#pragma unroll
        for (int it = 0; it < 10; ++it) {
            int e = tid + (it << 8);
            int kr = e / 80, gc = (e % 80) << 2;
            int j = kq + kr; if (j > n - 1) j = n - 1;
            fb[it] = *(const float4*)(Wh + (size_t)j * D + gc);
        }
    };
    auto storeTile = [&](int buf, int kq) {
#pragma unroll
        for (int it = 0; it < 2; ++it) {
            int j = kq + pcol[it];
            float p0 = 0.f, p1 = 0.f;
            if (j <= n - 2) {
                if (pcnt[it] == 0) { p0 = invn; p1 = invn; }
                else {
                    if (a2r[it].x > 0.f)
                        p0 = __expf(lrelu(pw1[it] + w2r[it].x) - prm[it]) * prz[it];
                    if (a2r[it].y > 0.f)
                        p1 = __expf(lrelu(pw1[it] + w2r[it].y) - prm[it]) * prz[it];
                }
            }
            Pack2 pk; pk.b[0] = (__bf16)p0; pk.b[1] = (__bf16)p1;
            *(unsigned int*)(Ps[buf] + prow[it] * 32 + pcol[it]) = pk.u32;
        }
#pragma unroll
        for (int it = 0; it < 10; ++it) {
            int e = tid + (it << 8);
            int kr = e / 80, gc = (e % 80) << 2;
            BsT[buf][(gc + 0) * 32 + kr] = (__bf16)fb[it].x;
            BsT[buf][(gc + 1) * 32 + kr] = (__bf16)fb[it].y;
            BsT[buf][(gc + 2) * 32 + kr] = (__bf16)fb[it].z;
            BsT[buf][(gc + 3) * 32 + kr] = (__bf16)fb[it].w;
        }
    };

    v8f acc[5] = {};
    loadRegs(0);
    storeTile(0, 0);
    for (int kk = 0; kk < n; kk += 32) {
        const int cur = (kk >> 5) & 1;
        __syncthreads();
        const bool more = (kk + 32 < n);
        if (more) loadRegs(kk + 32);
        const v16bf af = *(const v16bf*)(Ps[cur] + (mhalf * 16 + mrow) * 32 + hi * 16);
#pragma unroll
        for (int j = 0; j < 5; ++j) {
            const v16bf bfm = *(const v16bf*)(BsT[cur] + (ncol0 + j * 16 + mrow) * 32 + hi * 16);
            acc[j] = __builtin_amdgcn_wmma_f32_16x16x32_bf16(false, af, false, bfm,
                                                             (short)0, acc[j], false, false);
        }
        if (more) storeTile(cur ^ 1, kk + 32);
    }
#pragma unroll
    for (int j = 0; j < 5; ++j) {
        int c = ncol0 + j * 16 + mrow;
#pragma unroll
        for (int r = 0; r < 8; ++r) {
            int m = row0 + mhalf * 16 + r + hi * 8;
            if (m < n) {
                float v = acc[j][r];
                if (act) v = elu1(v);
                if (resid) v += resid[(size_t)m * D + c];
                out[(size_t)m * D + c] = v;
            }
        }
    }
}

// ---------------- pooling / unpooling ----------------
__global__ void pool_score_kernel(const float* __restrict__ X, const float* __restrict__ w,
                                  const float* __restrict__ bptr, float* __restrict__ scores, int n) {
    int row  = blockIdx.x * (blockDim.x >> 5) + (threadIdx.x >> 5);
    int lane = threadIdx.x & 31;
    if (row >= n) return;
    const float* xr = X + (size_t)row * D;
    float s = 0.f;
    for (int c = lane; c < D; c += 32) s += xr[c] * w[c];
    for (int off = 16; off > 0; off >>= 1) s += __shfl_xor(s, off, 32);
    if (lane == 0) {
        float t = (s + bptr[0]) * 0.01f;
        scores[row] = 1.f / (1.f + __expf(-t));
    }
}

// exact stable top-k by rank counting (matches jax.lax.top_k ordering)
__global__ void topk_kernel(const float* __restrict__ s, int n, int nkeep,
                            int* __restrict__ idx, float* __restrict__ vals) {
    int i = blockIdx.x * blockDim.x + threadIdx.x;
    if (i >= n) return;
    float si = s[i];
    int rank = 0;
    for (int j = 0; j < n; ++j) {
        float sj = s[j];
        if (sj > si || (sj == si && j < i)) rank++;
    }
    if (rank < nkeep) { idx[rank] = i; vals[rank] = si; }
}

__global__ void gather_rows_kernel(const float* __restrict__ X, const int* __restrict__ idx,
                                   const float* __restrict__ vals, float* __restrict__ out, int nk) {
    int t = blockIdx.x * blockDim.x + threadIdx.x;
    if (t >= nk * D) return;
    int r = t / D, c = t % D;
    out[t] = X[(size_t)idx[r] * D + c] * vals[r];
}

__global__ void gather_adj_kernel(const float* __restrict__ A, const int* __restrict__ idx,
                                  float* __restrict__ out, int nold, int nk) {
    int c = blockIdx.x * blockDim.x + threadIdx.x;
    int r = blockIdx.y;
    if (c >= nk) return;
    out[(size_t)r * nk + c] = A[(size_t)idx[r] * nold + idx[c]];
}

__global__ void gate_kernel(const float* __restrict__ X, const float* __restrict__ G,
                            float* __restrict__ out, int total) {
    int t = blockIdx.x * blockDim.x + threadIdx.x;
    if (t < total) out[t] = X[t] * (1.f / (1.f + __expf(-G[t])));
}

__global__ void scatter_rows_kernel(const float* __restrict__ src, const int* __restrict__ idx,
                                    float* __restrict__ dst, int nk) {
    int t = blockIdx.x * blockDim.x + threadIdx.x;
    if (t >= nk * D) return;
    int r = t / D, c = t % D;
    dst[(size_t)idx[r] * D + c] = src[t];
}

__global__ void concat_kernel(const float* __restrict__ Xa, const float* __restrict__ Xb,
                              float* __restrict__ out, int n) {
    int t = blockIdx.x * blockDim.x + threadIdx.x;
    if (t >= n * 2 * D) return;
    int i = t / (2 * D), c = t % (2 * D);
    out[t] = (c < D) ? Xa[(size_t)i * D + c] : Xb[(size_t)i * D + (c - D)];
}

// ---------------- host orchestration ----------------
static inline int cdiv(int a, int b) { return (a + b - 1) / b; }

struct GatScratch {
    float *Wh, *Wh1, *Wh2, *rowM, *rowZinv;
    int* rowCnt;
};

static void run_gat(hipStream_t st, const float* adj, int n, const float* h, int kin,
                    const float* W, const float* avec, float* out, const float* resid,
                    int act, const GatScratch& sc) {
    wmma_gemm_kernel<<<cdiv(n, 32), 256, 0, st>>>(h, W, nullptr, sc.Wh, n, kin, 0);
    gemv2_kernel<<<cdiv(n, 4), 128, 0, st>>>(sc.Wh, avec, sc.Wh1, sc.Wh2, n);
    attn_stats_kernel<<<cdiv(n, 4), 128, 0, st>>>(adj, sc.Wh1, sc.Wh2,
                                                  sc.rowM, sc.rowZinv, sc.rowCnt, n);
    attn_apply_kernel<<<cdiv(n, 32), 256, 0, st>>>(adj, sc.Wh, sc.Wh1, sc.Wh2,
                                                   sc.rowM, sc.rowZinv, sc.rowCnt,
                                                   out, resid, n, act);
}

extern "C" void kernel_launch(void* const* d_in, const int* in_sizes, int n_in,
                              void* d_out, int out_size, void* d_ws, size_t ws_size,
                              hipStream_t stream) {
    (void)in_sizes; (void)n_in; (void)out_size; (void)ws_size;
    const float* A        = (const float*)d_in[0];
    const float* X        = (const float*)d_in[1];
    const float* start_W  = (const float*)d_in[2];
    const float* start_a  = (const float*)d_in[3];
    const float* bottom_W = (const float*)d_in[4];
    const float* bottom_a = (const float*)d_in[5];
    const float* end_W    = (const float*)d_in[6];
    const float* end_a    = (const float*)d_in[7];
    const float* down_W   = (const float*)d_in[8];
    const float* down_a   = (const float*)d_in[9];
    const float* up_W     = (const float*)d_in[10];
    const float* up_a     = (const float*)d_in[11];
    const float* pool_w   = (const float*)d_in[12];
    const float* pool_b   = (const float*)d_in[13];
    const float* unpool_w = (const float*)d_in[14];
    const float* unpool_b = (const float*)d_in[15];

    float* Xout     = (float*)d_out;               // [N0 * D]
    float* startOut = Xout + (size_t)N0 * D;       // [N0 * D]

    static const int ns[5] = {4096, 3686, 2580, 1548, 774};  // level sizes

    float* w = (float*)d_ws;
    size_t off = 0;
    auto alloc = [&](size_t cnt) { float* p = w + off; off += cnt; return p; };

    float* A1 = alloc((size_t)ns[1] * ns[1]);
    float* A2 = alloc((size_t)ns[2] * ns[2]);
    float* A3 = alloc((size_t)ns[3] * ns[3]);
    float* A4 = alloc((size_t)ns[4] * ns[4]);
    const float* adj[5] = {A, A1, A2, A3, A4};
    float* adjw[5] = {nullptr, A1, A2, A3, A4};

    float* vbuf = alloc(N0);
    float* ybuf = alloc(N0);
    float* ss   = alloc(4);
    float* Xc   = alloc((size_t)N0 * D);
    float* orgX = alloc((size_t)N0 * D);
    float* bufA = alloc((size_t)N0 * D);
    float* bufB = alloc((size_t)N0 * D);

    GatScratch sc;
    sc.Wh      = alloc((size_t)N0 * D);
    sc.Wh1     = alloc(N0);
    sc.Wh2     = alloc(N0);
    sc.rowM    = alloc(N0);
    sc.rowZinv = alloc(N0);
    sc.rowCnt  = (int*)alloc(N0);

    float* downs[4];
    for (int i = 0; i < 4; ++i) downs[i] = alloc((size_t)ns[i] * D);
    int*   idxs[4];
    float* vals[4];
    for (int i = 0; i < 4; ++i) {
        idxs[i] = (int*)alloc(ns[i + 1]);
        vals[i] = alloc(ns[i + 1]);
    }
    float* scores = alloc(N0);
    float* G      = alloc((size_t)N0 * D);
    float* gated  = alloc((size_t)N0 * D);
    float* Xcat   = alloc((size_t)N0 * 2 * D);

    // --- 1. eigenvector centrality: 30 power iterations (A L2-resident) ---
    init_v_kernel<<<cdiv(N0, 256), 256, 0, stream>>>(vbuf, N0);
    for (int it = 0; it < 30; ++it) {
        gemv_kernel<<<cdiv(N0, 8), 256, 0, stream>>>(A, vbuf, ybuf, N0);
        sumsq_kernel<<<1, 256, 0, stream>>>(ybuf, ss, N0);
        normalize_kernel<<<cdiv(N0, 256), 256, 0, stream>>>(ybuf, ss, vbuf, N0);
    }

    // --- 2. Xc = [X | centrality], start GAT -> orgX (also second output) ---
    build_xc_kernel<<<cdiv(N0 * D, 256), 256, 0, stream>>>(X, vbuf, Xc, N0);
    run_gat(stream, A, N0, Xc, D, start_W, start_a, orgX, nullptr, 1, sc);

    // --- 3. down path: GAT + top-k pool per level ---
    float* cur = orgX;
    for (int i = 0; i < 4; ++i) {
        run_gat(stream, adj[i], ns[i], cur, D,
                down_W + (size_t)i * D * D, down_a + (size_t)i * 2 * D,
                downs[i], nullptr, 1, sc);
        pool_score_kernel<<<cdiv(ns[i], 4), 128, 0, stream>>>(downs[i], pool_w + (size_t)i * D,
                                                              pool_b + i, scores, ns[i]);
        topk_kernel<<<cdiv(ns[i], 256), 256, 0, stream>>>(scores, ns[i], ns[i + 1],
                                                          idxs[i], vals[i]);
        float* nxt = (cur == bufA) ? bufB : bufA;
        gather_rows_kernel<<<cdiv(ns[i + 1] * D, 256), 256, 0, stream>>>(downs[i], idxs[i],
                                                                         vals[i], nxt, ns[i + 1]);
        gather_adj_kernel<<<dim3(cdiv(ns[i + 1], 256), ns[i + 1]), 256, 0, stream>>>(
            adj[i], idxs[i], adjw[i + 1], ns[i], ns[i + 1]);
        cur = nxt;
    }

    // --- 4. bottom GAT at n=774 ---
    {
        float* out = (cur == bufA) ? bufB : bufA;
        run_gat(stream, adj[4], ns[4], cur, D, bottom_W, bottom_a, out, nullptr, 1, sc);
        cur = out;
    }

    // --- 5. up path: gated unpool (scatter) + GAT + skip add ---
    for (int i = 0; i < 4; ++i) {
        int upi = 3 - i;
        int nsm = ns[upi + 1], nbg = ns[upi];
        float* other = (cur == bufA) ? bufB : bufA;
        // G = cur @ unpool_w[i]^T + unpool_b[i]   (WMMA, transposed B access)
        wmma_gemm_kernel<<<cdiv(nsm, 32), 256, 0, stream>>>(
            cur, unpool_w + (size_t)i * D * D, unpool_b + (size_t)i * D, G, nsm, D, 1);
        gate_kernel<<<cdiv(nsm * D, 256), 256, 0, stream>>>(cur, G, gated, nsm * D);
        zero_kernel<<<cdiv(nbg * D, 256), 256, 0, stream>>>(other, nbg * D);
        scatter_rows_kernel<<<cdiv(nsm * D, 256), 256, 0, stream>>>(gated, idxs[upi], other, nsm);
        // GAT at big size, residual = down_outs[upi] (out reuses cur's buffer)
        run_gat(stream, adj[upi], nbg, other, D,
                up_W + (size_t)i * D * D, up_a + (size_t)i * 2 * D,
                cur, downs[upi], 1, sc);
    }

    // --- 6. end: concat [Xh | orgX] (640), final GAT (no ELU) -> Xout; copy start outs ---
    concat_kernel<<<cdiv(N0 * 2 * D, 256), 256, 0, stream>>>(cur, orgX, Xcat, N0);
    run_gat(stream, A, N0, Xcat, 2 * D, end_W, end_a, Xout, nullptr, 0, sc);
    hipMemcpyAsync(startOut, orgX, (size_t)N0 * D * sizeof(float),
                   hipMemcpyDeviceToDevice, stream);
}
